// Retention_13975823582269
// MI455X (gfx1250) — matrix operand order repaired and checked by reading, not compiled, not run on hardware
//
#include <hip/hip_runtime.h>
#include <hip/hip_bf16.h>

// MI455X / gfx1250 retention kernel.
// Workspace layout (needs 3 * 8*2048*768 * 2B = 75.5 MB):
//   Qb [16384][768] bf16, Kb [16384][768] bf16, Vt [8][768][2048] bf16 (V transposed)

typedef __attribute__((ext_vector_type(16))) __bf16 v16bf;
typedef __attribute__((ext_vector_type(8)))  __bf16 v8bf;
typedef __attribute__((ext_vector_type(2)))  __bf16 v2bf;
typedef __attribute__((ext_vector_type(8)))  float  v8f;
typedef __attribute__((ext_vector_type(4)))  float  v4f;

static constexpr int BS = 8;
static constexpr int SL = 2048;
static constexpr int D  = 768;
static constexpr int N3 = 3 * D;           // 2304

static constexpr float L2A    = -0.014499569695115089f; // log2(0.99)
static constexpr float RSQRTD = 0.036084391824351615f;  // 1/sqrt(768)

__device__ __forceinline__ v16bf cat16(v8bf lo, v8bf hi) {
  return __builtin_shufflevector(lo, hi, 0,1,2,3,4,5,6,7,8,9,10,11,12,13,14,15);
}
__device__ __forceinline__ v8f wmma_bf16(v16bf a, v16bf b, v8f c) {
  // D = A(16x32 bf16) * B(32x16 bf16) + C(16x16 f32)
  return __builtin_amdgcn_wmma_f32_16x16x32_bf16(false, a, false, b, (short)0, c,
                                                 false, false);
}
__device__ __forceinline__ v8f zero8() {
  v8f z = {0.f,0.f,0.f,0.f,0.f,0.f,0.f,0.f};
  return z;
}

// ---------------------------------------------------------------------------
// Kernel 1: QKV = x @ W + b, split + convert to bf16 (V stored transposed).
// Grid (N3/256, M/128), block 256 (8 waves). Wave tile 64x64 (4x4 WMMA tiles).
// ---------------------------------------------------------------------------
__global__ __launch_bounds__(256)
void qkv_proj_kernel(const float* __restrict__ x, const float* __restrict__ W,
                     const float* __restrict__ bq,
                     __bf16* __restrict__ Qb, __bf16* __restrict__ Kb,
                     __bf16* __restrict__ Vt) {
  // W tile staged transposed: ldsw[col 0..255][kk 0..31], row stride 56 (16B-aligned)
  __shared__ alignas(16) __bf16 ldsw[256 * 56];

  const int tid  = threadIdx.x;
  const int lane = tid & 31;
  const int w    = tid >> 5;
  const int lh   = lane >> 4;   // lane half (0/1)
  const int ll   = lane & 15;
  const int wgN  = blockIdx.x * 256;
  const int wgM  = blockIdx.y * 128;
  const int waveM = (w >> 2) * 64;
  const int waveN = (w & 3) * 64;

  v8f acc[4][4];
#pragma unroll
  for (int mt = 0; mt < 4; ++mt)
#pragma unroll
    for (int nt = 0; nt < 4; ++nt) acc[mt][nt] = zero8();

#pragma unroll 1
  for (int dc = 0; dc < D; dc += 32) {
    __syncthreads();
    // cooperative stage: W[dc..dc+31][wgN..wgN+255] -> bf16, transposed in LDS
    {
      const int c0  = (tid & 63) * 4;     // 4 consecutive columns
      const int kk0 = (tid >> 6) * 8;     // 8 consecutive rows
#pragma unroll
      for (int r = 0; r < 4; ++r) {
        const int kkA = kk0 + 2 * r;
        const float* wr = W + (size_t)(dc + kkA) * N3 + wgN + c0;
        v4f a = *(const v4f*)wr;          // row kkA
        v4f b = *(const v4f*)(wr + N3);   // row kkA+1
#pragma unroll
        for (int c = 0; c < 4; ++c) {
          v2bf p;
          p.x = (__bf16)a[c];
          p.y = (__bf16)b[c];
          *(v2bf*)&ldsw[(c0 + c) * 56 + kkA] = p;   // packed (kk, kk+1)
        }
      }
    }
    __syncthreads();

    // A fragments: x rows, f32 -> bf16 on the fly
    v16bf af[4];
#pragma unroll
    for (int mt = 0; mt < 4; ++mt) {
      const int row = wgM + waveM + mt * 16 + ll;
      const float* xp = x + (size_t)row * D + dc + lh * 8;
      v4f f0 = *(const v4f*)xp;
      v4f f1 = *(const v4f*)(xp + 4);
      v4f f2 = *(const v4f*)(xp + 16);
      v4f f3 = *(const v4f*)(xp + 20);
      v16bf a;
#pragma unroll
      for (int c = 0; c < 4; ++c) {
        a[c]      = (__bf16)f0[c];
        a[4 + c]  = (__bf16)f1[c];
        a[8 + c]  = (__bf16)f2[c];
        a[12 + c] = (__bf16)f3[c];
      }
      af[mt] = a;
    }

    // B fragments from LDS (contiguous after transpose) + 16 WMMAs
#pragma unroll
    for (int nt = 0; nt < 4; ++nt) {
      const int colL = waveN + nt * 16 + ll;
      const __bf16* lp = &ldsw[colL * 56 + lh * 8];
      v16bf bf = cat16(*(const v8bf*)lp, *(const v8bf*)(lp + 16));
#pragma unroll
      for (int mt = 0; mt < 4; ++mt)
        acc[mt][nt] = wmma_bf16(af[mt], bf, acc[mt][nt]);
    }
  }

  // epilogue: bias, convert, split to Q / K / V^T
#pragma unroll
  for (int nt = 0; nt < 4; ++nt) {
    const int col = wgN + waveN + nt * 16 + ll;
    const float bias = bq[col];
#pragma unroll
    for (int mt = 0; mt < 4; ++mt) {
#pragma unroll
      for (int r = 0; r < 8; ++r) {
        const int row = wgM + waveM + mt * 16 + r + lh * 8; // C layout
        const float v = acc[mt][nt][r] + bias;
        if (col < D) {
          Qb[(size_t)row * D + col] = (__bf16)v;
        } else if (col < 2 * D) {
          Kb[(size_t)row * D + (col - D)] = (__bf16)v;
        } else {
          const int bidx = row >> 11;          // batch (2048 rows each)
          const int jin  = row & (SL - 1);
          Vt[((size_t)bidx * D + (col - 2 * D)) * SL + jin] = (__bf16)v;
        }
      }
    }
  }
}

// ---------------------------------------------------------------------------
// Kernel 2: out[i] = sum_{j<=i} alpha^(i-j)/sqrt(d) * (q_i . k_j) * v_j
// Grid (SL/64, BS), block 256 (8 waves). WG owns 64 i-rows.
// Wave w: i_sub = w>>1 (16 rows), half = w&1 (phase A: j-tile pair,
//                                            phase B: 384-col out half).
// ---------------------------------------------------------------------------
__global__ __launch_bounds__(256)
void retention_kernel(const __bf16* __restrict__ Qb, const __bf16* __restrict__ Kb,
                      const __bf16* __restrict__ Vt, float* __restrict__ out) {
  __shared__ alignas(16) __bf16 sS[64 * 72];  // S tile 64x64 bf16, stride 72

  const int tid  = threadIdx.x;
  const int lane = tid & 31;
  const int w    = tid >> 5;
  const int lh   = lane >> 4;
  const int ll   = lane & 15;
  const int b    = blockIdx.y;
  const int I    = blockIdx.x * 64;
  const int isub = w >> 1;
  const int half = w & 1;

  // preload q A-fragments for this wave's 16 rows across full depth (24 x v16bf)
  v16bf qf[24];
  {
    const int row = b * SL + I + isub * 16 + ll;
    const __bf16* qp = Qb + (size_t)row * D + lh * 8;
#pragma unroll
    for (int t = 0; t < 24; ++t) {
      const __bf16* p = qp + t * 32;
      qf[t] = cat16(*(const v8bf*)p, *(const v8bf*)(p + 16));
    }
  }

  v8f o[24];
#pragma unroll
  for (int nt = 0; nt < 24; ++nt) o[nt] = zero8();

  const int nj = I / 64 + 1;   // causal: only j-blocks up to the diagonal
  for (int js = 0; js < nj; ++js) {
    const int J = js * 64;

    // ---- Phase A: S = (q k^T) * alpha^(i-j)/sqrt(d), into LDS as bf16 ----
#pragma unroll
    for (int jtl = 0; jtl < 2; ++jtl) {
      const int jt  = half * 2 + jtl;
      const int ii  = I + isub * 16;
      const int jjb = J + jt * 16;
      if (jjb <= ii + 15) {                       // wave-uniform liveness
        v8f s = zero8();
        const int jrow = b * SL + jjb + ll;
        const __bf16* kp = Kb + (size_t)jrow * D + lh * 8;
#pragma unroll
        for (int t = 0; t < 24; ++t) {            // K = 768 in steps of 32
          const __bf16* p = kp + t * 32;
          v16bf kb = cat16(*(const v8bf*)p, *(const v8bf*)(p + 16));
          s = wmma_bf16(qf[t], kb, s);
        }
#pragma unroll
        for (int r = 0; r < 8; ++r) {             // mask + store (C layout)
          const int i = ii + r + lh * 8;
          const int j = jjb + ll;
          const float m =
              (i >= j) ? __builtin_amdgcn_exp2f((float)(i - j) * L2A) * RSQRTD
                       : 0.f;
          sS[(i - I) * 72 + (j - J)] = (__bf16)(s[r] * m);
        }
      } else {                                    // fully above diagonal
#pragma unroll
        for (int r = 0; r < 8; ++r) {
          const int i = ii + r + lh * 8;
          sS[(i - I) * 72 + (jjb - J) + ll] = (__bf16)0.f;
        }
      }
    }
    __syncthreads();

    // ---- Phase B: out += S @ V (V^T gives contiguous B fragments) ----
#pragma unroll
    for (int jjs = 0; jjs < 2; ++jjs) {
      const int jj = jjs * 32;
      const __bf16* sp = &sS[(isub * 16 + ll) * 72 + jj + lh * 8];
      v16bf sa = cat16(*(const v8bf*)sp, *(const v8bf*)(sp + 16));
#pragma unroll
      for (int nt = 0; nt < 24; ++nt) {
        const int ncol = half * 384 + nt * 16 + ll;
        const __bf16* vp = Vt + ((size_t)b * D + ncol) * SL + J + jj + lh * 8;
        v16bf vb = cat16(*(const v8bf*)vp, *(const v8bf*)(vp + 16));
        o[nt] = wmma_bf16(sa, vb, o[nt]);
      }
    }
    __syncthreads();
  }

  // store out (f32)
#pragma unroll
  for (int nt = 0; nt < 24; ++nt) {
#pragma unroll
    for (int r = 0; r < 8; ++r) {
      const int i   = I + isub * 16 + r + lh * 8;
      const int col = half * 384 + nt * 16 + ll;
      out[((size_t)(b * SL + i)) * D + col] = o[nt][r];
    }
  }
}

// ---------------------------------------------------------------------------
extern "C" void kernel_launch(void* const* d_in, const int* in_sizes, int n_in,
                              void* d_out, int out_size, void* d_ws, size_t ws_size,
                              hipStream_t stream) {
  const float* x  = (const float*)d_in[0];   // [8,2048,768]
  const float* Wq = (const float*)d_in[1];   // [768,2304]
  const float* bq = (const float*)d_in[2];   // [2304]
  float* out = (float*)d_out;                // [8,2048,768]

  __bf16* Qb = (__bf16*)d_ws;
  __bf16* Kb = Qb + (size_t)BS * SL * D;
  __bf16* Vt = Kb + (size_t)BS * SL * D;

  qkv_proj_kernel<<<dim3(N3 / 256, (BS * SL) / 128), 256, 0, stream>>>(
      x, Wq, bq, Qb, Kb, Vt);
  retention_kernel<<<dim3(SL / 64, BS), 256, 0, stream>>>(Qb, Kb, Vt, out);
}
